// TGATModel_73512660238835
// MI455X (gfx1250) — matrix operand order repaired and checked by reading, not compiled, not run on hardware
//
#include <hip/hip_runtime.h>
#include <math.h>

// ---------------- problem constants ----------------
constexpr int N_NODES = 50000;
constexpr int E_EDGES = 1600000;
constexpr int IN_DIM  = 128;
constexpr int HID     = 64;
constexpr int HEADS   = 4;
constexpr int HD      = 16;
constexpr int NP      = 50048;   // 391 * 128, multiple of 128 (8 waves x 16 rows)
constexpr int KMAX    = 192;     // largest K (layer 0)

typedef __attribute__((ext_vector_type(16))) _Float16 v16h;
typedef __attribute__((ext_vector_type(8)))  _Float16 v8h;
typedef __attribute__((ext_vector_type(8)))  float    v8f;

// ---------------- helpers ----------------
__device__ __forceinline__ unsigned enc_f32(float f) {
    unsigned u = __float_as_uint(f);
    return (u & 0x80000000u) ? ~u : (u | 0x80000000u);
}
__device__ __forceinline__ float dec_f32(unsigned e) {
    unsigned u = (e & 0x80000000u) ? (e & 0x7fffffffu) : ~e;
    return __uint_as_float(u);
}
// enc(-inf) = 0x007FFFFF
#define ENC_NEG_INF 0x007FFFFFu

// ---------------- kernel 1: build A = [feat | cos(t*omega+phase)] in f16 ----------------
// A: [NP x K] f16 row-major. Rows >= n_valid are zero (padding for WMMA tiles).
__global__ void build_A_kernel(_Float16* __restrict__ A,
                               const float* __restrict__ feat, int in_dim,
                               const float* __restrict__ time,
                               const float* __restrict__ omega,
                               const float* __restrict__ phase,
                               int K, int n_valid) {
    long long t = (long long)blockIdx.x * blockDim.x + threadIdx.x;
    long long total = (long long)NP * K;
    if (t >= total) return;
    int n = (int)(t / K);
    int k = (int)(t - (long long)n * K);
    float v = 0.f;
    if (n < n_valid) {
        if (k < in_dim) v = feat[(long long)n * in_dim + k];
        else {
            int j = k - in_dim;
            v = __cosf(time[n] * omega[j] + phase[j]);
        }
    }
    A[t] = (_Float16)v;
}

// ---------------- kernel 2: repack W (f32 [K x 64]) into fragment-major f16 ----------------
// Bpack element index t = ((ks*4 + nt)*32 + lane)*16 + e
// maps to W[kk*64 + col], kk = ks*32 + (lane>>4)*16 + e, col = nt*16 + (lane&15)
__global__ void pack_W_kernel(_Float16* __restrict__ Bpack,
                              const float* __restrict__ W, int K) {
    int t = blockIdx.x * blockDim.x + threadIdx.x;
    if (t >= K * 64) return;
    int e    = t & 15;
    int lane = (t >> 4) & 31;
    int nt   = (t >> 9) & 3;
    int ks   = t >> 11;
    int kk   = ks * 32 + ((lane >> 4) << 4) + e;
    int col  = nt * 16 + (lane & 15);
    Bpack[t] = (_Float16)W[kk * 64 + col];
}

// ---------------- kernel 3: WMMA GEMM  C[NP x 64] = A[NP x K] * W[K x 64] ----------------
// 256 threads = 8 waves; wave w covers rows (blockIdx.x*8 + w)*16 .. +15, all 64 cols.
// B fragments staged once per block into LDS (<= 24KB of 320KB/WGP).
__global__ __launch_bounds__(256) void wmma_gemm_kernel(
                                 const _Float16* __restrict__ A,
                                 const _Float16* __restrict__ Bpack,
                                 float* __restrict__ C, int K) {
    __shared__ _Float16 Bs[KMAX * 64];         // fragment-major, 24576 B max

    // cooperative copy: K*128 bytes = K*8 chunks of 16B
    {
        const v8h* src = (const v8h*)Bpack;
        v8h*       dstl = (v8h*)Bs;
        int nchunks = K * 8;
        for (int c = threadIdx.x; c < nchunks; c += 256)
            dstl[c] = src[c];
    }
    __syncthreads();

    int wave = threadIdx.x >> 5;
    int lane = threadIdx.x & 31;
    int mbase = (blockIdx.x * 8 + wave) * 16;
    if (mbase >= NP) return;
    int row = lane & 15;
    int hi  = lane >> 4;           // 0: lanes 0-15, 1: lanes 16-31

    v8f acc0 = {}, acc1 = {}, acc2 = {}, acc3 = {};
    const v16h* Bp = (const v16h*)Bs;

    for (int kb = 0; kb < K; kb += 32) {
        // A fragment: lane<16 holds K = kb+{0..15}; lane>=16 holds kb+{16..31}
        const _Float16* Ap = A + (long long)(mbase + row) * K + kb + hi * 16;
        v8h a0 = *(const v8h*)(Ap);
        v8h a1 = *(const v8h*)(Ap + 8);
        v16h a;
#pragma unroll
        for (int i = 0; i < 8; ++i) { a[i] = a0[i]; a[8 + i] = a1[i]; }

        int fb = (kb >> 5) * 4;
        v16h b0 = Bp[(fb + 0) * 32 + lane];
        v16h b1 = Bp[(fb + 1) * 32 + lane];
        v16h b2 = Bp[(fb + 2) * 32 + lane];
        v16h b3 = Bp[(fb + 3) * 32 + lane];

        acc0 = __builtin_amdgcn_wmma_f32_16x16x32_f16(false, a, false, b0, (short)0, acc0, false, false);
        acc1 = __builtin_amdgcn_wmma_f32_16x16x32_f16(false, a, false, b1, (short)0, acc1, false, false);
        acc2 = __builtin_amdgcn_wmma_f32_16x16x32_f16(false, a, false, b2, (short)0, acc2, false, false);
        acc3 = __builtin_amdgcn_wmma_f32_16x16x32_f16(false, a, false, b3, (short)0, acc3, false, false);
    }

    // C/D layout: VGPR r -> row mbase + hi*8 + r, col = nt*16 + (lane&15)
    int col = lane & 15;
#pragma unroll
    for (int r = 0; r < 8; ++r) {
        long long rowoff = (long long)(mbase + hi * 8 + r) * 64;
        C[rowoff +  0 + col] = acc0[r];
        C[rowoff + 16 + col] = acc1[r];
        C[rowoff + 32 + col] = acc2[r];
        C[rowoff + 48 + col] = acc3[r];
    }
}

// ---------------- kernel 4: per-node attention scores ----------------
__global__ void node_scores_kernel(const float* __restrict__ h,
                                   const float* __restrict__ asrc,
                                   const float* __restrict__ adst,
                                   float* __restrict__ ssrc,
                                   float* __restrict__ sdst, int n_valid) {
    int t = blockIdx.x * blockDim.x + threadIdx.x;
    if (t >= n_valid * HEADS) return;
    int n  = t >> 2;
    int hd = t & 3;
    const float* hp = h + (long long)n * HID + hd * HD;
    float ss = 0.f, sd = 0.f;
#pragma unroll
    for (int j = 0; j < HD; ++j) {
        float v = hp[j];
        ss += v * asrc[hd * HD + j];
        sd += v * adst[hd * HD + j];
    }
    ssrc[t] = ss;
    sdst[t] = sd;
}

// ---------------- kernel 5: init softmax stats + accumulator ----------------
__global__ void init_accum_kernel(unsigned* __restrict__ menc,
                                  float* __restrict__ den,
                                  float* __restrict__ acc) {
    long long t = (long long)blockIdx.x * blockDim.x + threadIdx.x;
    if (t >= (long long)NP * HID) return;
    acc[t] = 0.f;
    int j = (int)(t & 63);
    if (j < HEADS) {
        long long idx = (t >> 6) * HEADS + j;
        menc[idx] = ENC_NEG_INF;
        den[idx]  = 0.f;
    }
}

// ---------------- kernel 6: segment max of edge scores (ordered-uint atomicMax) ----------------
__global__ void edge_max_kernel(const int* __restrict__ src,
                                const int* __restrict__ dst,
                                const float* __restrict__ ssrc,
                                const float* __restrict__ sdst,
                                unsigned* __restrict__ menc) {
    int e = blockIdx.x * blockDim.x + threadIdx.x;
    if (e >= E_EDGES) return;
    int s = src[e], d = dst[e];
#pragma unroll
    for (int hd = 0; hd < HEADS; ++hd) {
        float sc = ssrc[s * HEADS + hd] + sdst[d * HEADS + hd];
        sc = sc > 0.f ? sc : 0.2f * sc;               // leaky_relu(0.2)
        atomicMax(&menc[d * HEADS + hd], enc_f32(sc));
    }
}

// ---------------- kernel 7: softmax numerators + weighted aggregation ----------------
// one thread per (edge, head): 1 den atomic + 16 acc atomics
__global__ void edge_accum_kernel(const int* __restrict__ src,
                                  const int* __restrict__ dst,
                                  const float* __restrict__ ew,
                                  const float* __restrict__ ssrc,
                                  const float* __restrict__ sdst,
                                  const unsigned* __restrict__ menc,
                                  const float* __restrict__ h,
                                  float* __restrict__ den,
                                  float* __restrict__ acc) {
    long long t = (long long)blockIdx.x * blockDim.x + threadIdx.x;
    if (t >= (long long)E_EDGES * HEADS) return;
    int e  = (int)(t >> 2);
    int hd = (int)(t & 3);
    int s = src[e], d = dst[e];
    float sc = ssrc[s * HEADS + hd] + sdst[d * HEADS + hd];
    sc = sc > 0.f ? sc : 0.2f * sc;
    float m   = dec_f32(menc[d * HEADS + hd]);
    float num = expf(sc - m) * ew[e];
    atomicAdd(&den[d * HEADS + hd], num);
    const float* hp = h + (long long)s * HID + hd * HD;
    float* ap       = acc + (long long)d * HID + hd * HD;
#pragma unroll
    for (int j = 0; j < HD; ++j)
        atomicAdd(&ap[j], num * hp[j]);
}

// ---------------- kernel 8: normalize + ELU (layer-0 output) ----------------
__global__ void finalize_elu_kernel(const float* __restrict__ acc,
                                    const float* __restrict__ den,
                                    float* __restrict__ hout) {
    long long t = (long long)blockIdx.x * blockDim.x + threadIdx.x;
    if (t >= (long long)NP * HID) return;
    int n = (int)(t >> 6);
    int j = (int)(t & 63);
    float v = 0.f;
    if (n < N_NODES) {
        v = acc[t] / (den[n * HEADS + (j >> 4)] + 1e-16f);
        v = v > 0.f ? v : expm1f(v);                  // elu(alpha=1)
    }
    hout[t] = v;                                      // pad rows zeroed
}

// ---------------- kernel 9: normalize + linear head ----------------
__global__ void finalize_head_kernel(const float* __restrict__ acc,
                                     const float* __restrict__ den,
                                     const float* __restrict__ Whead,
                                     const float* __restrict__ bhead,
                                     float* __restrict__ out) {
    int n = blockIdx.x * blockDim.x + threadIdx.x;
    if (n >= N_NODES) return;
    float sum = bhead[0];
#pragma unroll
    for (int j = 0; j < HID; ++j) {
        float v = acc[(long long)n * HID + j] / (den[n * HEADS + (j >> 4)] + 1e-16f);
        sum += v * Whead[j];
    }
    out[n] = sum;
}

// ---------------- host-side orchestration ----------------
static inline int cdiv(long long a, int b) { return (int)((a + b - 1) / b); }

extern "C" void kernel_launch(void* const* d_in, const int* in_sizes, int n_in,
                              void* d_out, int out_size, void* d_ws, size_t ws_size,
                              hipStream_t stream) {
    const float* x      = (const float*)d_in[0];
    const int*   ei     = (const int*)  d_in[1];
    const float* ew     = (const float*)d_in[2];
    const float* time   = (const float*)d_in[3];
    const float* omega0 = (const float*)d_in[4];
    const float* phase0 = (const float*)d_in[5];
    const float* W0     = (const float*)d_in[6];
    const float* asrc0  = (const float*)d_in[7];
    const float* adst0  = (const float*)d_in[8];
    const float* omega1 = (const float*)d_in[9];
    const float* phase1 = (const float*)d_in[10];
    const float* W1     = (const float*)d_in[11];
    const float* asrc1  = (const float*)d_in[12];
    const float* adst1  = (const float*)d_in[13];
    const float* Whead  = (const float*)d_in[14];
    const float* bhead  = (const float*)d_in[15];
    float* out = (float*)d_out;

    const int* src = ei;
    const int* dst = ei + E_EDGES;

    // ---- workspace sub-allocation (256B aligned) ----
    char* base = (char*)d_ws;
    size_t off = 0;
    auto take = [&](size_t bytes) -> char* {
        char* p = base + off;
        off = (off + bytes + 255) & ~(size_t)255;
        return p;
    };
    _Float16* A     = (_Float16*)take((size_t)NP * KMAX * sizeof(_Float16));
    _Float16* Wf    = (_Float16*)take((size_t)KMAX * 64 * sizeof(_Float16));
    float*    h     = (float*)   take((size_t)NP * HID * sizeof(float));
    float*    ssrc  = (float*)   take((size_t)NP * HEADS * sizeof(float));
    float*    sdst  = (float*)   take((size_t)NP * HEADS * sizeof(float));
    unsigned* menc  = (unsigned*)take((size_t)NP * HEADS * sizeof(unsigned));
    float*    den   = (float*)   take((size_t)NP * HEADS * sizeof(float));
    float*    acc   = (float*)   take((size_t)NP * HID * sizeof(float));
    float*    hout  = (float*)   take((size_t)NP * HID * sizeof(float));
    (void)ws_size; (void)in_sizes; (void)n_in; (void)out_size;

    const int B = 256;
    const int gemm_grid = NP / 128;   // 8 waves x 16 rows per block

    // =================== layer 0 (K = 192) ===================
    build_A_kernel<<<cdiv((long long)NP * 192, B), B, 0, stream>>>(
        A, x, IN_DIM, time, omega0, phase0, 192, N_NODES);
    pack_W_kernel<<<cdiv(192 * 64, B), B, 0, stream>>>(Wf, W0, 192);
    wmma_gemm_kernel<<<gemm_grid, B, 0, stream>>>(A, Wf, h, 192);
    node_scores_kernel<<<cdiv(N_NODES * HEADS, B), B, 0, stream>>>(
        h, asrc0, adst0, ssrc, sdst, N_NODES);
    init_accum_kernel<<<cdiv((long long)NP * HID, B), B, 0, stream>>>(menc, den, acc);
    edge_max_kernel<<<cdiv(E_EDGES, B), B, 0, stream>>>(src, dst, ssrc, sdst, menc);
    edge_accum_kernel<<<cdiv((long long)E_EDGES * HEADS, B), B, 0, stream>>>(
        src, dst, ew, ssrc, sdst, menc, h, den, acc);
    finalize_elu_kernel<<<cdiv((long long)NP * HID, B), B, 0, stream>>>(acc, den, hout);

    // =================== layer 1 (K = 128) ===================
    build_A_kernel<<<cdiv((long long)NP * 128, B), B, 0, stream>>>(
        A, hout, HID, time, omega1, phase1, 128, N_NODES);
    pack_W_kernel<<<cdiv(128 * 64, B), B, 0, stream>>>(Wf, W1, 128);
    wmma_gemm_kernel<<<gemm_grid, B, 0, stream>>>(A, Wf, h, 128);
    node_scores_kernel<<<cdiv(N_NODES * HEADS, B), B, 0, stream>>>(
        h, asrc1, adst1, ssrc, sdst, N_NODES);
    init_accum_kernel<<<cdiv((long long)NP * HID, B), B, 0, stream>>>(menc, den, acc);
    edge_max_kernel<<<cdiv(E_EDGES, B), B, 0, stream>>>(src, dst, ssrc, sdst, menc);
    edge_accum_kernel<<<cdiv((long long)E_EDGES * HEADS, B), B, 0, stream>>>(
        src, dst, ew, ssrc, sdst, menc, h, den, acc);

    // =================== linear head ===================
    finalize_head_kernel<<<cdiv(N_NODES, B), B, 0, stream>>>(
        acc, den, Whead, bhead, out);
}
